// MultiHeadAttention_7490422964285
// MI455X (gfx1250) — matrix operand order repaired and checked by reading, not compiled, not run on hardware
//
#include <hip/hip_runtime.h>
#include <hip/hip_bf16.h>
#include <math.h>
#include <stdint.h>

// ---------------- problem constants (fixed by the reference) ----------------
#define BATCH 4
#define SEQ   2048
#define DIM   1024
#define HEADS 16
#define DH    64            // DIM / HEADS
#define MROWS (BATCH * SEQ) // 8192 rows for the projection GEMMs
#define QT    16            // query rows per fused-attention block
#define LDS_STRIDE 2056     // 2048 + 8 pad floats (break 64-bank periodicity)
#define VCHUNK 64           // K rows of V staged per TDM/LDS chunk

typedef __attribute__((ext_vector_type(16))) _Float16 v16h;
typedef __attribute__((ext_vector_type(8)))  _Float16 v8h;
typedef __attribute__((ext_vector_type(8)))  float    v8f;
typedef __attribute__((ext_vector_type(4)))  unsigned int u32x4;
typedef __attribute__((ext_vector_type(4)))  int      i32x4;
typedef __attribute__((ext_vector_type(8)))  int      i32x8;

// ---------------------------------------------------------------------------
// CDNA5 WMMA 16x16x32 f16 fragment layouts as used here (contiguous k, same order for A and B)
//
// A (16x32, MxK): lane l, m = l&15, hi = l>>4.
//   halves j=0..7  -> K = hi*16 + j         (contiguous 8)
//   halves j=8..15 -> K = hi*16 + 8 + (j-8) (contiguous 8)
// B (32x16, KxN): lane l, n = l&15, hi = l>>4.
//   halves j=0..15 -> K = hi*16 + j         (contiguous 16)
// C/D (16x16 f32): vgpr r, lane l -> element (m = r + 8*hi, n = l&15)
// ---------------------------------------------------------------------------

__device__ inline v16h frag_a_from_f32(const float* __restrict__ p, int hi) {
  const int base = hi * 16;
  v16h a;
#pragma unroll
  for (int j = 0; j < 8; ++j) a[j] = (_Float16)p[base + j];
#pragma unroll
  for (int j = 0; j < 8; ++j) a[8 + j] = (_Float16)p[base + 8 + j];
  return a;
}

__device__ inline v16h frag_a_from_f16(const _Float16* __restrict__ p, int hi) {
  v8h lo = *(const v8h*)(p + hi * 16);
  v8h hiv = *(const v8h*)(p + hi * 16 + 8);
  v16h a;
#pragma unroll
  for (int j = 0; j < 8; ++j) { a[j] = lo[j]; a[8 + j] = hiv[j]; }
  return a;
}

__device__ inline v16h frag_b_from_f32(const float* __restrict__ p) {
  v16h b;
#pragma unroll
  for (int j = 0; j < 16; ++j) b[j] = (_Float16)p[j];
  return b;
}

// Issue a TDM DMA: 64-row x 64-elem f16 tile (row stride DIM elems) -> LDS.
// D# packing per CDNA5 ISA (group0 128b, group1 256b).
// Probe result (round 4): this toolchain ships the therock TDM header and the
// 6-arg builtin (uint32x4, int32x8, int32x4, int32x4, int32x8, i32).
__device__ inline void tdm_load_v_chunk(const _Float16* gsrc,
                                        unsigned lds_byte_off) {
  unsigned long long ga = (unsigned long long)(uintptr_t)gsrc;
  u32x4 g0;
  g0[0] = 1u;                                  // count=1, user mode
  g0[1] = lds_byte_off;                        // lds_addr (bytes)
  g0[2] = (unsigned)ga;                        // global_addr[31:0]
  g0[3] = (unsigned)((ga >> 32) & 0x01FFFFFFull) | (2u << 30); // [56:32]|type=2
  i32x8 g1;
  g1[0] = (int)(1u << 16);     // workgroup_mask=0, data_size=1 (2 bytes)
  g1[1] = (int)(64u << 16);    // tensor_dim0 = 64 (bits 79:48, low half)
  g1[2] = (int)(64u << 16);    // tensor_dim0 hi=0 | tensor_dim1 = 64 (low)
  g1[3] = (int)(64u << 16);    // tensor_dim1 hi=0 | tile_dim0 = 64
  g1[4] = VCHUNK;              // tile_dim1 = 64, tile_dim2 = 0
  g1[5] = (int)DIM;            // tensor_dim0_stride = 1024 elems (bits 207:160)
  g1[6] = 0;                   // dim0_stride hi | tensor_dim1_stride lo
  g1[7] = 0;
  i32x4 z4 = {0, 0, 0, 0};
#if __has_include(<hip/amd_detail/amd_gfx1250_TDM.h>)
  i32x8 z8 = {0, 0, 0, 0, 0, 0, 0, 0};
  __builtin_amdgcn_tensor_load_to_lds(g0, g1, z4, z4, z8, 0);  // 6-arg form
#else
  __builtin_amdgcn_tensor_load_to_lds(g0, g1, z4, z4, 0);      // 5-arg form
#endif
}

// ---------------------------------------------------------------------------
// Kernel 1: Y = relu(X @ W^T + bias), X f32 [M,K], W f32 [N,K], Y f16 [M,N]
// ---------------------------------------------------------------------------
__global__ void gemm_proj_kernel(const float* __restrict__ X,
                                 const float* __restrict__ W,
                                 const float* __restrict__ bias,
                                 _Float16* __restrict__ Y,
                                 int M, int N, int K) {
  const int wave = (int)((blockIdx.x * blockDim.x + threadIdx.x) >> 5);
  const int lane = threadIdx.x & 31;
  const int l16 = lane & 15, hi = lane >> 4;
  const int tilesN = N >> 4;
  const int numTiles = (M >> 4) * tilesN;
  if (wave >= numTiles) return;
  const int tm = wave / tilesN, tn = wave % tilesN;
  const int m0 = tm << 4, n0 = tn << 4;

  const float* xrow = X + (size_t)(m0 + l16) * K;
  const float* wrow = W + (size_t)(n0 + l16) * K;

  v8f c = {};
  for (int k0 = 0; k0 < K; k0 += 32) {
    __builtin_prefetch(xrow + k0 + 64, 0, 3);   // global_prefetch_b8
    __builtin_prefetch(wrow + k0 + 64, 0, 3);
    v16h a = frag_a_from_f32(xrow + k0, hi);
    v16h b = frag_b_from_f32(wrow + k0 + hi * 16);
    c = __builtin_amdgcn_wmma_f32_16x16x32_f16(false, a, false, b,
                                               (short)0, c, false, false);
  }
  const float bn = bias[n0 + l16];
#pragma unroll
  for (int r = 0; r < 8; ++r) {
    float v = fmaxf(c[r] + bn, 0.0f);
    const int m = m0 + r + 8 * hi;
    Y[(size_t)m * N + (n0 + l16)] = (_Float16)v;
  }
}

// ---------------------------------------------------------------------------
// Kernel 2 (fused): per (b, h, 16-query tile):
//   scores = Qh Kh^T / 8  -> LDS (16 x 2048 f32 strip, resident)
//   softmax rows in LDS, single coalesced write of probs to d_out attn region
//   y_tile = probs @ Vh with V chunks DMA'd into LDS by the TDM
// 8 waves / 256 threads. LDS ~157 KB (ISA: up to 320 KB per workgroup).
// ---------------------------------------------------------------------------
__global__ __launch_bounds__(256) void fused_attn_kernel(
    const _Float16* __restrict__ qh, const _Float16* __restrict__ kh,
    const _Float16* __restrict__ vh, float* __restrict__ attn,
    _Float16* __restrict__ yi) {
  __shared__ float sc[QT * LDS_STRIDE];        // 131584 B score strip
  __shared__ _Float16 vbuf[2][VCHUNK * DH];    // 16384 B V double buffer
  __shared__ float redbuf[256];
  __shared__ float rowmax[QT];
  __shared__ float rowsum[QT];
  __shared__ float psum[2][QT][DH];            // K-split partials for attn @ V

  const int blk = blockIdx.x;
  const int qt = blk & ((SEQ >> 4) - 1);  // 128 query tiles
  const int bh = blk >> 7;
  const int b = bh / HEADS, h = bh % HEADS;
  const int qbase = qt << 4;

  const int tid = threadIdx.x;
  const int wave = tid >> 5;
  const int lane = tid & 31;
  const int l16 = lane & 15, hi = lane >> 4;

  // ---- Phase 1: scores -> LDS. Q fragments loaded once, reused 16x. ----
  const _Float16* qrow = qh + (size_t)(b * SEQ + qbase + l16) * DIM + h * DH;
  const v16h a0 = frag_a_from_f16(qrow, hi);
  const v16h a1 = frag_a_from_f16(qrow + 32, hi);
  const float scale = 0.125f;  // 1/sqrt(64)

  for (int i = 0; i < 16; ++i) {
    const int tn = wave * 16 + i;  // key tile 0..127
    const _Float16* krow =
        kh + (size_t)(b * SEQ + (tn << 4) + l16) * DIM + h * DH;
    v16h b0 = *(const v16h*)(krow + hi * 16);
    v16h b1 = *(const v16h*)(krow + 32 + hi * 16);
    v8f c = {};
    c = __builtin_amdgcn_wmma_f32_16x16x32_f16(false, a0, false, b0,
                                               (short)0, c, false, false);
    c = __builtin_amdgcn_wmma_f32_16x16x32_f16(false, a1, false, b1,
                                               (short)0, c, false, false);
#pragma unroll
    for (int r = 0; r < 8; ++r)
      sc[(r + 8 * hi) * LDS_STRIDE + (tn << 4) + l16] = c[r] * scale;
  }
  __syncthreads();

  // ---- Phase 2: softmax rows in LDS (16 threads per row). ----
  const int row = tid >> 4;
  const int sub = tid & 15;
  float mx = -INFINITY;
  for (int col = sub; col < SEQ; col += 16)
    mx = fmaxf(mx, sc[row * LDS_STRIDE + col]);
  redbuf[tid] = mx;
  __syncthreads();
  for (int s = 8; s > 0; s >>= 1) {
    if (sub < s) redbuf[tid] = fmaxf(redbuf[tid], redbuf[tid + s]);
    __syncthreads();
  }
  if (sub == 0) rowmax[row] = redbuf[tid];
  __syncthreads();
  mx = rowmax[row];

  float sum = 0.0f;
  for (int col = sub; col < SEQ; col += 16) {
    float e = __expf(sc[row * LDS_STRIDE + col] - mx);
    sc[row * LDS_STRIDE + col] = e;
    sum += e;
  }
  redbuf[tid] = sum;
  __syncthreads();
  for (int s = 8; s > 0; s >>= 1) {
    if (sub < s) redbuf[tid] += redbuf[tid + s];
    __syncthreads();
  }
  if (sub == 0) rowsum[row] = redbuf[tid];
  __syncthreads();

  // Normalize in LDS + the ONLY global pass over the attn matrix (coalesced).
  for (int idx = tid; idx < QT * SEQ; idx += 256) {
    const int rr = idx >> 11;          // / 2048
    const int col = idx & (SEQ - 1);
    const float vv = sc[rr * LDS_STRIDE + col] * (1.0f / rowsum[rr]);
    sc[rr * LDS_STRIDE + col] = vv;
    attn[((size_t)(b * SEQ + qbase + rr) * HEADS + h) * SEQ + col] = vv;
  }
  __syncthreads();

  // ---- Phase 3: y_tile = probs @ Vh, V staged through LDS in 64-row chunks.
  // 8 waves = 4 DH-tiles x 2 K-substeps per chunk; partials in psum.
  {
    const int tn = wave & 3;    // DH tile (16 cols each)
    const int ks = wave >> 2;   // K substep within a chunk
    const int dcol = (tn << 4) + l16;
    const size_t vbase = (size_t)b * SEQ * DIM + h * DH;
    v8f c = {};

    // Double-buffered TDM pipeline: wave 0 drives the DMA engine.
    if (wave == 0)
      tdm_load_v_chunk(vh + vbase, (unsigned)(uintptr_t)&vbuf[0][0]);
    for (int ch = 0; ch < SEQ / VCHUNK; ++ch) {
      if (wave == 0) {
        if (ch + 1 < SEQ / VCHUNK) {
          tdm_load_v_chunk(vh + vbase + (size_t)(ch + 1) * VCHUNK * DIM,
                           (unsigned)(uintptr_t)&vbuf[(ch + 1) & 1][0]);
          __builtin_amdgcn_s_wait_tensorcnt((short)1);  // chunk ch complete
        } else {
          __builtin_amdgcn_s_wait_tensorcnt((short)0);
        }
      }
      __syncthreads();  // publish chunk ch
      const _Float16* vb = &vbuf[ch & 1][0];
      const int k0 = ch * VCHUNK + ks * 32;
      const float* ap = &sc[l16 * LDS_STRIDE + k0];
      v16h a;
#pragma unroll
      for (int j = 0; j < 8; ++j) a[j] = (_Float16)ap[hi * 16 + j];
#pragma unroll
      for (int j = 0; j < 8; ++j) a[8 + j] = (_Float16)ap[hi * 16 + 8 + j];
      v16h bf;
#pragma unroll
      for (int j = 0; j < 16; ++j)
        bf[j] = vb[(ks * 32 + hi * 16 + j) * DH + dcol];
      c = __builtin_amdgcn_wmma_f32_16x16x32_f16(false, a, false, bf,
                                                 (short)0, c, false, false);
      __syncthreads();  // all reads done before buffer reuse
    }

#pragma unroll
    for (int r = 0; r < 8; ++r) psum[ks][r + 8 * hi][dcol] = c[r];
  }
  __syncthreads();

  // Combine K-halves, store y tile as f16.
  for (int idx = tid; idx < QT * DH; idx += 256) {
    const int rr = idx >> 6;   // / 64
    const int dd = idx & 63;
    const float vv = psum[0][rr][dd] + psum[1][rr][dd];
    yi[(size_t)(b * SEQ + qbase + rr) * DIM + h * DH + dd] = (_Float16)vv;
  }
}

// ---------------------------------------------------------------------------
// Kernel 3: out = relu(Yi @ Wo^T + bo), Yi f16 [M,K], Wo f32 [N,K], out f32
// ---------------------------------------------------------------------------
__global__ void gemm_out_kernel(const _Float16* __restrict__ Yi,
                                const float* __restrict__ W,
                                const float* __restrict__ bias,
                                float* __restrict__ out,
                                int M, int N, int K) {
  const int wave = (int)((blockIdx.x * blockDim.x + threadIdx.x) >> 5);
  const int lane = threadIdx.x & 31;
  const int l16 = lane & 15, hi = lane >> 4;
  const int tilesN = N >> 4;
  const int numTiles = (M >> 4) * tilesN;
  if (wave >= numTiles) return;
  const int tm = wave / tilesN, tn = wave % tilesN;
  const int m0 = tm << 4, n0 = tn << 4;

  const _Float16* yrow = Yi + (size_t)(m0 + l16) * K;
  const float* wrow = W + (size_t)(n0 + l16) * K;

  v8f c = {};
  for (int k0 = 0; k0 < K; k0 += 32) {
    __builtin_prefetch(wrow + k0 + 64, 0, 3);
    v16h a = frag_a_from_f16(yrow + k0, hi);
    v16h b = frag_b_from_f32(wrow + k0 + hi * 16);
    c = __builtin_amdgcn_wmma_f32_16x16x32_f16(false, a, false, b,
                                               (short)0, c, false, false);
  }
  const float bn = bias[n0 + l16];
#pragma unroll
  for (int r = 0; r < 8; ++r) {
    float v = fmaxf(c[r] + bn, 0.0f);
    const int m = m0 + r + 8 * hi;
    out[(size_t)m * N + (n0 + l16)] = v;
  }
}

// ---------------------------------------------------------------------------
extern "C" void kernel_launch(void* const* d_in, const int* in_sizes, int n_in,
                              void* d_out, int out_size, void* d_ws,
                              size_t ws_size, hipStream_t stream) {
  const float* q  = (const float*)d_in[0];
  const float* k  = (const float*)d_in[1];
  const float* v  = (const float*)d_in[2];
  const float* Wq = (const float*)d_in[3];
  const float* bq = (const float*)d_in[4];
  const float* Wk = (const float*)d_in[5];
  const float* bk = (const float*)d_in[6];
  const float* Wv = (const float*)d_in[7];
  const float* bv = (const float*)d_in[8];
  const float* Wo = (const float*)d_in[9];
  const float* bo = (const float*)d_in[10];

  // d_out layout: y [B,S,D] f32, then attn [B,S,H,S] f32
  float* out_y = (float*)d_out;
  float* out_attn = out_y + (size_t)BATCH * SEQ * DIM;

  // workspace: qh/kh/vh/yi as f16 [B*S, DIM] (4 x 16 MiB = 64 MiB)
  const size_t NE = (size_t)BATCH * SEQ * DIM;
  _Float16* qh = (_Float16*)d_ws;
  _Float16* kh = qh + NE;
  _Float16* vh = kh + NE;
  _Float16* yi = vh + NE;

  const dim3 blk(256);

  // Projections: 32768 tiles / 8 waves per block = 4096 blocks
  const int projBlocks = ((MROWS >> 4) * (DIM >> 4)) / 8;
  gemm_proj_kernel<<<projBlocks, blk, 0, stream>>>(q, Wq, bq, qh, MROWS, DIM, DIM);
  gemm_proj_kernel<<<projBlocks, blk, 0, stream>>>(k, Wk, bk, kh, MROWS, DIM, DIM);
  gemm_proj_kernel<<<projBlocks, blk, 0, stream>>>(v, Wv, bv, vh, MROWS, DIM, DIM);

  // Fused attention: one block per (b, h, 16-query tile) = 8192 blocks
  fused_attn_kernel<<<BATCH * HEADS * (SEQ >> 4), blk, 0, stream>>>(
      qh, kh, vh, out_attn, yi);

  // Output projection
  gemm_out_kernel<<<projBlocks, blk, 0, stream>>>(yi, Wo, bo, out_y, MROWS, DIM, DIM);
}
